// AligNet_27401891349265
// MI455X (gfx1250) — matrix operand order repaired and checked by reading, not compiled, not run on hardware
//
#include <hip/hip_runtime.h>
#include <hip/hip_bf16.h>

typedef __attribute__((ext_vector_type(16))) _Float16 v16h;
typedef __attribute__((ext_vector_type(8)))  _Float16 v8h;
typedef __attribute__((ext_vector_type(8)))  float    v8f;

#define NFCH 48
#define HH 96
#define WW 96
#define HWP 9216            // H*W
#define NB 2
#define NPIX 18432          // NB*H*W
#define KDIM 432            // 48*9
#define KP 448              // K padded to multiple of 32
#define NKSTEP 14           // KP/32
#define MP 5120             // 5040 padded to 40*128
#define MREAL 5040
#define FEAT_ELEMS 884736   // NB*NFCH*HWP

// ---- workspace layout (bytes) ----
#define OFF_WP    0u
#define OFF_BIAS  4587520u      // MP*KP*2
#define OFF_BP    4608000u      // + MP*4
#define OFF_OM    21123072u     // + NPIX*KP*2
#define OFF_FEATS 209866752u    // + MP*NPIX*2
#define OFF_S     220483584u    // + 3*FEAT_ELEMS*4
#define OFF_ATT   220483968u    // + 96*4
#define OFF_WD5   220485120u    // + 288*4 (rounded)
#define OFF_WD3   220601856u    // + 48*1216*2
#define OFF_WD1   220644864u    // + 48*448*2  (end: +48*64*2 = 220651008)

// ---------------------------------------------------------------------------
// 1) Pack the three offset/mask conv weight sets into one f16 [MP][KP] matrix
//    (K order: ci*9 + ky*3 + kx, matching im2row), plus f32 bias vector.
// ---------------------------------------------------------------------------
__global__ void __launch_bounds__(256)
pack_weights_kernel(const float* __restrict__ w5, const float* __restrict__ b5,
                    const float* __restrict__ w3, const float* __restrict__ b3,
                    const float* __restrict__ w1, const float* __restrict__ b1,
                    _Float16* __restrict__ Wp, float* __restrict__ bias)
{
    int idx = blockIdx.x * 256 + threadIdx.x;
    if (idx >= MP * KP) return;
    int m = idx / KP, k = idx - m * KP;
    float v = 0.f;
    if (m < MREAL && k < KDIM) {
        int ci = k / 9, r = k - ci * 9;
        const float* w; int co;
        if (m < 3600)      { w = w5; co = m; }
        else if (m < 4896) { w = w3; co = m - 3600; }
        else               { w = w1; co = m - 4896; }
        v = w[((co * NFCH + ci) * 3 + (r / 3)) * 3 + (r % 3)];
    }
    Wp[idx] = (_Float16)v;
    if (k == 0) {
        float bv = 0.f;
        if (m < 3600)       bv = b5[m];
        else if (m < 4896)  bv = b3[m - 3600];
        else if (m < MREAL) bv = b1[m - 4896];
        bias[m] = bv;
    }
}

// Pack deformable-conv weights: Wd[o][ck] = w[o][c][k], ck = c*K + k, zero-pad.
__global__ void __launch_bounds__(256)
pack_dcw_kernel(const float* __restrict__ w, _Float16* __restrict__ Wd, int CK, int CKP)
{
    int idx = blockIdx.x * 256 + threadIdx.x;
    if (idx >= NFCH * CKP) return;
    int o = idx / CKP, ck = idx - o * CKP;
    float v = (ck < CK) ? w[(size_t)o * CK + ck] : 0.f;
    Wd[idx] = (_Float16)v;
}

// ---------------------------------------------------------------------------
// 2) im2row of `out` (shared by all three 3x3 convs): f16 [NPIX][KP]
// ---------------------------------------------------------------------------
__global__ void __launch_bounds__(256)
im2row_kernel(const float* __restrict__ src, _Float16* __restrict__ Bp)
{
    int idx = blockIdx.x * 256 + threadIdx.x;
    if (idx >= NPIX * KP) return;
    int n = idx / KP, k = idx - n * KP;
    float v = 0.f;
    if (k < KDIM) {
        int ci = k / 9, r = k - ci * 9;
        int ky = r / 3, kx = r - ky * 3;
        int b = n / HWP, rem = n - b * HWP;
        int y = rem / WW, x = rem - y * WW;
        int yy = y + ky - 1, xx = x + kx - 1;
        if (yy >= 0 && yy < HH && xx >= 0 && xx < WW)
            v = src[((size_t)(b * NFCH + ci) * HH + yy) * WW + xx];
    }
    Bp[idx] = (_Float16)v;
}

// ---------------------------------------------------------------------------
// 3) WMMA GEMM: C[MP][NPIX] (f16, +bias) = Wp[MP][KP] * Bp[NPIX][KP]^T
//    256 threads (8 waves), 128x128 block tile, 32x64 wave tile,
//    double-buffered LDS (one barrier per K-step).
// ---------------------------------------------------------------------------
__global__ void __launch_bounds__(256)
gemm_wmma_kernel(const _Float16* __restrict__ A, const _Float16* __restrict__ B,
                 const float* __restrict__ bias, _Float16* __restrict__ C)
{
    __shared__ __align__(64) _Float16 As[2][128 * 32];
    __shared__ __align__(64) _Float16 Bs[2][128 * 32];
    const int tid   = threadIdx.x;
    const int lane  = tid & 31;
    const int waveM = (tid >> 5) >> 1;   // 0..3
    const int waveN = (tid >> 5) & 1;    // 0..1
    const int m0 = blockIdx.y * 128;
    const int n0 = blockIdx.x * 128;

    // staging: 2 float4 chunks per array per thread
    const int q0 = tid,       r0 = q0 >> 2, s0 = q0 & 3;
    const int q1 = tid + 256, r1 = q1 >> 2, s1 = q1 & 3;
    const _Float16* Ag0 = A + (size_t)(m0 + r0) * KP + s0 * 8;
    const _Float16* Ag1 = A + (size_t)(m0 + r1) * KP + s1 * 8;
    const _Float16* Bg0 = B + (size_t)(n0 + r0) * KP + s0 * 8;
    const _Float16* Bg1 = B + (size_t)(n0 + r1) * KP + s1 * 8;

    float4 ra0 = *(const float4*)(Ag0);
    float4 ra1 = *(const float4*)(Ag1);
    float4 rb0 = *(const float4*)(Bg0);
    float4 rb1 = *(const float4*)(Bg1);
    *(float4*)(&As[0][r0 * 32 + s0 * 8]) = ra0;
    *(float4*)(&As[0][r1 * 32 + s1 * 8]) = ra1;
    *(float4*)(&Bs[0][r0 * 32 + s0 * 8]) = rb0;
    *(float4*)(&Bs[0][r1 * 32 + s1 * 8]) = rb1;
    __syncthreads();

    v8f acc[2][4] = {};
    const int kh = (lane >> 4) * 8;
    const int kb = (lane >> 4) * 16;

    for (int kt = 0; kt < NKSTEP; ++kt) {
        const int cur = kt & 1;
        if (kt + 1 < NKSTEP) {
            const int ko = (kt + 1) * 32;
            ra0 = *(const float4*)(Ag0 + ko);
            ra1 = *(const float4*)(Ag1 + ko);
            rb0 = *(const float4*)(Bg0 + ko);
            rb1 = *(const float4*)(Bg1 + ko);
        }
        // A fragments (ISA 16-bit A layout: runs [kh..kh+7], [16+kh..+7])
        v16h af[2];
        #pragma unroll
        for (int mi = 0; mi < 2; ++mi) {
            const _Float16* ap = &As[cur][(waveM * 32 + mi * 16 + (lane & 15)) * 32];
            v8h alo = *(const v8h*)(ap + kh);
            v8h ahi = *(const v8h*)(ap + 16 + kh);
            #pragma unroll
            for (int v = 0; v < 8; ++v) { af[mi][v] = alo[v]; af[mi][8 + v] = ahi[v]; }
        }
        #pragma unroll
        for (int nt = 0; nt < 4; ++nt) {
            const _Float16* bp = &Bs[cur][(waveN * 64 + nt * 16 + (lane & 15)) * 32];
            v8h blo = *(const v8h*)(bp + kh);
            v8h bhi = *(const v8h*)(bp + 16 + kh);
            v16h bf;
            #pragma unroll
            for (int v = 0; v < 8; ++v) { bf[v] = blo[v]; bf[8 + v] = bhi[v]; }
            acc[0][nt] = __builtin_amdgcn_wmma_f32_16x16x32_f16(
                false, af[0], false, bf, (short)0, acc[0][nt], false, false);
            acc[1][nt] = __builtin_amdgcn_wmma_f32_16x16x32_f16(
                false, af[1], false, bf, (short)0, acc[1][nt], false, false);
        }
        if (kt + 1 < NKSTEP) {
            const int nxt = cur ^ 1;
            *(float4*)(&As[nxt][r0 * 32 + s0 * 8]) = ra0;
            *(float4*)(&As[nxt][r1 * 32 + s1 * 8]) = ra1;
            *(float4*)(&Bs[nxt][r0 * 32 + s0 * 8]) = rb0;
            *(float4*)(&Bs[nxt][r1 * 32 + s1 * 8]) = rb1;
        }
        __syncthreads();
    }

    // C layout per ISA: VGPR r, lane L -> m = r + (L>>4)*8, n = L&15
    const int rbase = m0 + waveM * 32 + ((lane >> 4) << 3);
    const int cbase = n0 + waveN * 64 + (lane & 15);
    #pragma unroll
    for (int mi = 0; mi < 2; ++mi) {
        #pragma unroll
        for (int nt = 0; nt < 4; ++nt) {
            #pragma unroll
            for (int r = 0; r < 8; ++r) {
                int m = rbase + mi * 16 + r;
                int n = cbase + nt * 16;
                C[(size_t)m * NPIX + n] = (_Float16)(acc[mi][nt][r] + bias[m]);
            }
        }
    }
}

// ---------------------------------------------------------------------------
// 4) Modulated deformable conv as WMMA GEMM over ck = c*K+k chunks.
//    Block = 128 threads, 64-pixel tile. Sampling phase builds LDS tile
//    st[64 px][32 ck] (coalesced om reads); WMMA phase: 3 M-tiles x 1 N-subtile
//    per wave, A-fragments straight from prepacked f16 weights (L2-hot).
// ---------------------------------------------------------------------------
__device__ __forceinline__ float bilin96(const float* __restrict__ p, float py, float px)
{
    float y0 = floorf(py), x0 = floorf(px);
    float acc = 0.f;
    #pragma unroll
    for (int dy = 0; dy < 2; ++dy) {
        #pragma unroll
        for (int dx = 0; dx < 2; ++dx) {
            float yi = y0 + (float)dy, xi = x0 + (float)dx;
            float wgt = (1.f - fabsf(py - yi)) * (1.f - fabsf(px - xi));
            bool valid = (yi >= 0.f) && (yi <= 95.f) && (xi >= 0.f) && (xi <= 95.f);
            int yc = min(max((int)yi, 0), 95);
            int xc = min(max((int)xi, 0), 95);
            acc += (valid ? wgt : 0.f) * p[yc * WW + xc];
        }
    }
    return acc;
}

template <int KS>
__global__ void __launch_bounds__(128)
deform_wmma_kernel(const _Float16* __restrict__ om, int rowBase,
                   const float* __restrict__ x, const _Float16* __restrict__ Wd,
                   const float* __restrict__ bvec, float* __restrict__ fout)
{
    constexpr int K   = KS * KS;
    constexpr int CK  = NFCH * K;
    constexpr int CKP = (CK + 31) & ~31;
    constexpr int NCH = CKP / 32;
    __shared__ __align__(64) _Float16 st[64 * 32];   // [px_local][ck_local]
    const int tid  = threadIdx.x;
    const int lane = tid & 31;
    const int wave = tid >> 5;          // pixel subtile 0..3
    const int n0 = blockIdx.x * 64;
    const int offRows = NFCH * 2 * K;

    v8f acc[3] = {};
    const int kh = (lane >> 4) * 8;
    const int kb = (lane >> 4) * 16;

    for (int ch = 0; ch < NCH; ++ch) {
        const int ck0 = ch * 32;
        __syncthreads();   // previous chunk's reads of st done
        // ---- sampling phase: 64 px x 32 ck = 2048 entries, 16 per thread ----
        #pragma unroll
        for (int i = 0; i < 16; ++i) {
            int item = tid + i * 128;
            int pxl = item & 63;        // pixels vary fastest -> coalesced om loads
            int ckl = item >> 6;
            int ck = ck0 + ckl;
            float sv = 0.f;
            if (ck < CK) {
                int c = ck / K, k = ck - c * K;
                int px = n0 + pxl;
                int b = px / HWP, rem = px - b * HWP;
                int y = rem / WW, xp = rem - y * WW;
                float oy = (float)om[(size_t)(rowBase + ck * 2 + 0) * NPIX + px];
                float ox = (float)om[(size_t)(rowBase + ck * 2 + 1) * NPIX + px];
                float mv = (float)om[(size_t)(rowBase + offRows + ck) * NPIX + px];
                float msk = 1.f / (1.f + __expf(-mv));
                float py  = (float)(y + k / KS - KS / 2) + oy;
                float pxx = (float)(xp + k % KS - KS / 2) + ox;
                sv = bilin96(x + (size_t)(b * NFCH + c) * HWP, py, pxx) * msk;
            }
            st[pxl * 32 + ckl] = (_Float16)sv;
        }
        __syncthreads();
        // ---- WMMA phase ----
        const _Float16* bp = &st[(wave * 16 + (lane & 15)) * 32];
        v8h blo = *(const v8h*)(bp + kh);
        v8h bhi = *(const v8h*)(bp + 16 + kh);
        v16h bf;
        #pragma unroll
        for (int v = 0; v < 8; ++v) { bf[v] = blo[v]; bf[8 + v] = bhi[v]; }
        #pragma unroll
        for (int mt = 0; mt < 3; ++mt) {
            const _Float16* apg = Wd + (size_t)(mt * 16 + (lane & 15)) * CKP + ck0;
            v8h alo = *(const v8h*)(apg + kh);
            v8h ahi = *(const v8h*)(apg + 16 + kh);
            v16h af;
            #pragma unroll
            for (int v = 0; v < 8; ++v) { af[v] = alo[v]; af[8 + v] = ahi[v]; }
            acc[mt] = __builtin_amdgcn_wmma_f32_16x16x32_f16(
                false, af, false, bf, (short)0, acc[mt], false, false);
        }
    }
    // epilogue: bias + ReLU
    const int px = n0 + wave * 16 + (lane & 15);
    const int b = px / HWP, rem = px - b * HWP;
    #pragma unroll
    for (int mt = 0; mt < 3; ++mt) {
        #pragma unroll
        for (int r = 0; r < 8; ++r) {
            int m = mt * 16 + ((lane >> 4) << 3) + r;
            fout[(size_t)(b * NFCH + m) * HWP + rem] = fmaxf(acc[mt][r] + bvec[m], 0.f);
        }
    }
}

// ---------------------------------------------------------------------------
// 5) SKFF: channel means, attention, combine
// ---------------------------------------------------------------------------
__global__ void __launch_bounds__(256)
skff_reduce_kernel(const float* __restrict__ feats, float* __restrict__ S)
{
    __shared__ float red[256];
    const int bc = blockIdx.x;     // 0..95
    float sum = 0.f;
    for (int p = threadIdx.x; p < HWP; p += 256) {
        size_t base = (size_t)bc * HWP + p;
        sum += feats[base] + feats[FEAT_ELEMS + base] + feats[2 * FEAT_ELEMS + base];
    }
    red[threadIdx.x] = sum;
    __syncthreads();
    for (int off = 128; off > 0; off >>= 1) {
        if (threadIdx.x < off) red[threadIdx.x] += red[threadIdx.x + off];
        __syncthreads();
    }
    if (threadIdx.x == 0) S[bc] = red[0] * (1.f / (float)HWP);
}

__global__ void __launch_bounds__(128)
skff_att_kernel(const float* __restrict__ S, const float* __restrict__ cw,
                const float* __restrict__ pa, const float* __restrict__ f0,
                const float* __restrict__ f1, const float* __restrict__ f2,
                float* __restrict__ att)
{
    __shared__ float Zs[12];
    int tid = threadIdx.x;
    if (tid < 12) {
        int b = tid / 6, r = tid - b * 6;
        float z = 0.f;
        for (int c = 0; c < NFCH; ++c) z += S[b * NFCH + c] * cw[r * NFCH + c];
        float a = *pa;
        Zs[tid] = (z >= 0.f) ? z : a * z;
    }
    __syncthreads();
    if (tid < 96) {
        int b = tid / NFCH, c = tid - b * NFCH;
        float L[3];
        const float* fw[3] = { f0, f1, f2 };
        #pragma unroll
        for (int j = 0; j < 3; ++j) {
            float t = 0.f;
            #pragma unroll
            for (int r = 0; r < 6; ++r) t += Zs[b * 6 + r] * fw[j][c * 6 + r];
            L[j] = t;
        }
        float mx = fmaxf(L[0], fmaxf(L[1], L[2]));
        float e0 = __expf(L[0] - mx), e1 = __expf(L[1] - mx), e2 = __expf(L[2] - mx);
        float inv = 1.f / (e0 + e1 + e2);
        att[(b * 3 + 0) * NFCH + c] = e0 * inv;
        att[(b * 3 + 1) * NFCH + c] = e1 * inv;
        att[(b * 3 + 2) * NFCH + c] = e2 * inv;
    }
}

__global__ void __launch_bounds__(256)
skff_combine_kernel(const float* __restrict__ feats, const float* __restrict__ att,
                    float* __restrict__ out)
{
    int idx = blockIdx.x * 256 + threadIdx.x;
    if (idx >= FEAT_ELEMS) return;
    int bc = idx / HWP;
    int b = bc / NFCH, c = bc - b * NFCH;
    float a0 = att[(b * 3 + 0) * NFCH + c];
    float a1 = att[(b * 3 + 1) * NFCH + c];
    float a2 = att[(b * 3 + 2) * NFCH + c];
    out[idx] = a0 * feats[idx] + a1 * feats[FEAT_ELEMS + idx] + a2 * feats[2 * FEAT_ELEMS + idx];
}

// ---------------------------------------------------------------------------
extern "C" void kernel_launch(void* const* d_in, const int* in_sizes, int n_in,
                              void* d_out, int out_size, void* d_ws, size_t ws_size,
                              hipStream_t stream)
{
    (void)in_sizes; (void)n_in; (void)out_size; (void)ws_size;
    const float* t_out = (const float*)d_in[0];
    const float* t_in  = (const float*)d_in[1];
    const float* om5_w = (const float*)d_in[2];
    const float* om5_b = (const float*)d_in[3];
    const float* om3_w = (const float*)d_in[4];
    const float* om3_b = (const float*)d_in[5];
    const float* om1_w = (const float*)d_in[6];
    const float* om1_b = (const float*)d_in[7];
    const float* dc5_w = (const float*)d_in[8];
    const float* dc5_b = (const float*)d_in[9];
    const float* dc3_w = (const float*)d_in[10];
    const float* dc3_b = (const float*)d_in[11];
    const float* dc1_w = (const float*)d_in[12];
    const float* dc1_b = (const float*)d_in[13];
    const float* sf_cw = (const float*)d_in[14];
    const float* sf_pa = (const float*)d_in[15];
    const float* sf_f0 = (const float*)d_in[16];
    const float* sf_f1 = (const float*)d_in[17];
    const float* sf_f2 = (const float*)d_in[18];

    char* ws = (char*)d_ws;
    _Float16* Wp    = (_Float16*)(ws + OFF_WP);
    float*    bias  = (float*)   (ws + OFF_BIAS);
    _Float16* Bp    = (_Float16*)(ws + OFF_BP);
    _Float16* om    = (_Float16*)(ws + OFF_OM);
    float*    feats = (float*)   (ws + OFF_FEATS);
    float*    Sbuf  = (float*)   (ws + OFF_S);
    float*    att   = (float*)   (ws + OFF_ATT);
    _Float16* Wd5   = (_Float16*)(ws + OFF_WD5);
    _Float16* Wd3   = (_Float16*)(ws + OFF_WD3);
    _Float16* Wd1   = (_Float16*)(ws + OFF_WD1);

    pack_weights_kernel<<<(MP * KP) / 256, 256, 0, stream>>>(
        om5_w, om5_b, om3_w, om3_b, om1_w, om1_b, Wp, bias);
    pack_dcw_kernel<<<(NFCH * 1216 + 255) / 256, 256, 0, stream>>>(dc5_w, Wd5, 1200, 1216);
    pack_dcw_kernel<<<(NFCH * 448 + 255) / 256, 256, 0, stream>>>(dc3_w, Wd3, 432, 448);
    pack_dcw_kernel<<<(NFCH * 64 + 255) / 256, 256, 0, stream>>>(dc1_w, Wd1, 48, 64);
    im2row_kernel<<<(NPIX * KP + 255) / 256, 256, 0, stream>>>(t_out, Bp);

    dim3 g(NPIX / 128, MP / 128);
    gemm_wmma_kernel<<<g, 256, 0, stream>>>(Wp, Bp, bias, om);

    // feats order matches reference stack([f1, f3, f5])
    deform_wmma_kernel<5><<<NPIX / 64, 128, 0, stream>>>(om, 0,    t_in, Wd5, dc5_b, feats + 2 * FEAT_ELEMS);
    deform_wmma_kernel<3><<<NPIX / 64, 128, 0, stream>>>(om, 3600, t_in, Wd3, dc3_b, feats + 1 * FEAT_ELEMS);
    deform_wmma_kernel<1><<<NPIX / 64, 128, 0, stream>>>(om, 4896, t_in, Wd1, dc1_b, feats + 0 * FEAT_ELEMS);

    skff_reduce_kernel<<<NB * NFCH, 256, 0, stream>>>(feats, Sbuf);
    skff_att_kernel<<<1, 128, 0, stream>>>(Sbuf, sf_cw, sf_pa, sf_f0, sf_f1, sf_f2, att);
    skff_combine_kernel<<<(FEAT_ELEMS + 255) / 256, 256, 0, stream>>>(feats, att, (float*)d_out);
}
